// Attention_16535624090434
// MI455X (gfx1250) — matrix operand order repaired and checked by reading, not compiled, not run on hardware
//
#include <hip/hip_runtime.h>

// ---------------------------------------------------------------------------
// MI455X (gfx1250, wave32) implementation of windowed attention block.
// Matmuls: v_wmma_f32_16x16x32_f16 (f16 in, f32 acc).
// Bias-table gather: TENSOR_LOAD_TO_LDS (TDM) + s_wait_tensorcnt.
// ---------------------------------------------------------------------------

typedef _Float16 v16h __attribute__((ext_vector_type(16)));
typedef _Float16 v8h  __attribute__((ext_vector_type(8)));
typedef float    v8f  __attribute__((ext_vector_type(8)));
typedef unsigned int u32x4 __attribute__((ext_vector_type(4)));
typedef int          i32x8 __attribute__((ext_vector_type(8)));
typedef int          i32x4 __attribute__((ext_vector_type(4)));

#define BATCH 128
#define NTOK  400          // T*V = 16*25
#define DIM   256
#define HEADS 8
#define HD    32           // head_dim == WMMA K for f16: one wmma per S tile
#define NPAD  416          // v^T row padding (multiple of 32)
#define NPOS  1519         // (2T-1)(2V-1) = 31*49
#define QSCALE 0.17677669529663687f   // 32^-0.5

static __device__ __forceinline__ v16h concat8(v8h lo, v8h hi) {
  v16h r;
#pragma unroll
  for (int i = 0; i < 8; ++i) { r[i] = lo[i]; r[i + 8] = hi[i]; }
  return r;
}

static __device__ __forceinline__ v8f wmma_f16(v16h a, v16h b, v8f c) {
  // (neg_a, A, neg_b, B, c_mod, C, reuse_a, reuse_b)
  return __builtin_amdgcn_wmma_f32_16x16x32_f16(false, a, false, b, (short)0, c,
                                                false, false);
}

// generic -> LDS(AS3) pointer, ptrtoint gives the 32-bit LDS byte offset
static __device__ __forceinline__ unsigned lds_byte_addr(void* p) {
  return (unsigned)(size_t)(__attribute__((address_space(3))) void*)p;
}

// ---------------------------------------------------------------------------
// Kernel 1: qkv = x @ qkv_w + qkv_b    (51200 x 256 x 768)
// Writes q (scaled) / k as f16 [B,H,N,32]; v transposed f16 [B,H,32,NPAD].
// Block: 256 thr (8 waves). Tile: M=128 N=64, K-chunks of 32.
// ---------------------------------------------------------------------------
__global__ __launch_bounds__(256)
void qkv_gemm_kernel(const float* __restrict__ x,
                     const float* __restrict__ w,
                     const float* __restrict__ bias,
                     _Float16* __restrict__ qh,
                     _Float16* __restrict__ kh,
                     _Float16* __restrict__ vT) {
  __shared__ __align__(16) _Float16 ldsA[128 * 40];  // [row][k], stride 40 halves
  __shared__ __align__(16) _Float16 ldsB[64 * 40];   // [col][k] (transposed)

  const int t    = threadIdx.x;
  const int lane = t & 31;
  const int wave = t >> 5;        // M strip 0..7
  const int ln16 = lane & 15;
  const int hi   = lane >> 4;     // 0: lanes 0-15, 1: lanes 16-31
  const int m0   = blockIdx.x * 128;
  const int n0   = blockIdx.y * 64;

  v8f acc[4] = {};

  for (int kk = 0; kk < DIM; kk += 32) {
    // --- stage A tile 128x32, f32 -> f16 ---
#pragma unroll
    for (int u = 0; u < 4; ++u) {
      int f   = u * 256 + t;           // 1024 float4 chunks
      int row = f >> 3;
      int c4  = (f & 7) * 4;
      const float4 v =
          *(const float4*)(x + (size_t)(m0 + row) * DIM + kk + c4);
      if (kk + 32 < DIM)  // L2 prefetch of next K chunk -> global_prefetch_b8
        __builtin_prefetch(x + (size_t)(m0 + row) * DIM + kk + 32 + c4, 0, 3);
      _Float16* dst = &ldsA[row * 40 + c4];
      dst[0] = (_Float16)v.x; dst[1] = (_Float16)v.y;
      dst[2] = (_Float16)v.z; dst[3] = (_Float16)v.w;
    }
    // --- stage B tile 32x64, transposed to [col][k], f32 -> f16 ---
    {
      int col = t & 63;
      int kg  = t >> 6;                // 0..3 -> k = kg*8 .. kg*8+7
      v8h pk;
#pragma unroll
      for (int k8 = 0; k8 < 8; ++k8)
        pk[k8] = (_Float16)w[(size_t)(kk + kg * 8 + k8) * 768 + n0 + col];
      *(v8h*)&ldsB[col * 40 + kg * 8] = pk;
    }
    __syncthreads();

    // --- fragments + WMMA ---
    const int arow = wave * 16 + ln16;
    const int kbA  = hi ? 16 : 0;      // A layout: lanes 16-31 hold K=16..31
    v16h afrag = concat8(*(const v8h*)&ldsA[arow * 40 + kbA],
                         *(const v8h*)&ldsA[arow * 40 + kbA + 8]);
    const int kbB = hi ? 16 : 0;       // B layout: lanes 16-31 hold K=16..31
#pragma unroll
    for (int j = 0; j < 4; ++j) {
      const int col = j * 16 + ln16;
      v16h bfrag = concat8(*(const v8h*)&ldsB[col * 40 + kbB],
                           *(const v8h*)&ldsB[col * 40 + kbB + 8]);
      acc[j] = wmma_f16(afrag, bfrag, acc[j]);
    }
    __syncthreads();
  }

  // --- epilogue: bias add + scatter to q / k / v^T ---
#pragma unroll
  for (int j = 0; j < 4; ++j) {
    const int c768 = n0 + j * 16 + ln16;
    const float bv = bias[c768];
    const int sel = c768 >> 8;          // 0=q 1=k 2=v
    const int hh  = (c768 >> 5) & 7;
    const int d   = c768 & 31;
#pragma unroll
    for (int r = 0; r < 8; ++r) {
      const int g = m0 + wave * 16 + r + hi * 8;  // global row (C layout)
      const int b = g / NTOK, n = g % NTOK;
      const float val = acc[j][r] + bv;
      if (sel == 0)
        qh[((size_t)(b * HEADS + hh) * NTOK + n) * HD + d] =
            (_Float16)(val * QSCALE);
      else if (sel == 1)
        kh[((size_t)(b * HEADS + hh) * NTOK + n) * HD + d] = (_Float16)val;
      else
        vT[((size_t)(b * HEADS + hh) * HD + d) * NPAD + n] = (_Float16)val;
    }
  }
}

// ---------------------------------------------------------------------------
// Kernel 2: dynamic position-bias MLP -> pos_tab[1519][8]  (tiny, scalar)
// BIASES row p = [ p/49 - 15 , p%49 - 24 ]
// ---------------------------------------------------------------------------
__global__ __launch_bounds__(256)
void posmlp_kernel(const float* __restrict__ pw,  const float* __restrict__ pb,
                   const float* __restrict__ g1,  const float* __restrict__ b1,
                   const float* __restrict__ w1,  const float* __restrict__ bb1,
                   const float* __restrict__ g2,  const float* __restrict__ b2,
                   const float* __restrict__ w2,  const float* __restrict__ bb2,
                   const float* __restrict__ g3,  const float* __restrict__ b3,
                   const float* __restrict__ w3,  const float* __restrict__ bb3,
                   float* __restrict__ pos_tab) {
  const int p = blockIdx.x * 256 + threadIdx.x;
  if (p >= NPOS) return;
  const float bh = (float)(p / 49 - 15);
  const float bw = (float)(p % 49 - 24);

  float h[16], y[16];
#pragma unroll
  for (int j = 0; j < 16; ++j) h[j] = bh * pw[j] + bw * pw[16 + j] + pb[j];

  const float* G[3]  = {g1, g2, g3};
  const float* Bt[3] = {b1, b2, b3};
  const float* W[2]  = {w1, w2};
  const float* Bw[2] = {bb1, bb2};

#pragma unroll
  for (int layer = 0; layer < 3; ++layer) {
    float mu = 0.f;
#pragma unroll
    for (int j = 0; j < 16; ++j) mu += h[j];
    mu *= (1.f / 16.f);
    float var = 0.f;
#pragma unroll
    for (int j = 0; j < 16; ++j) { float d = h[j] - mu; var += d * d; }
    var *= (1.f / 16.f);
    const float rs = rsqrtf(var + 1e-5f);
#pragma unroll
    for (int j = 0; j < 16; ++j) {
      float v = (h[j] - mu) * rs * G[layer][j] + Bt[layer][j];
      y[j] = v > 0.f ? v : 0.f;
    }
    if (layer < 2) {
#pragma unroll
      for (int j = 0; j < 16; ++j) {
        float s = Bw[layer][j];
#pragma unroll
        for (int k = 0; k < 16; ++k) s += y[k] * W[layer][k * 16 + j];
        h[j] = s;
      }
    }
  }
  // final 16 -> HEADS
#pragma unroll
  for (int hh = 0; hh < HEADS; ++hh) {
    float s = bb3[hh];
#pragma unroll
    for (int k = 0; k < 16; ++k) s += y[k] * w3[k * HEADS + hh];
    pos_tab[(size_t)p * HEADS + hh] = s;
  }
}

// ---------------------------------------------------------------------------
// Kernel 3: flash attention. One wave per (b, h, 16-row tile). 52 WMMAs/wave.
//   pos bias column (1519 floats, stride 32B) staged by the Tensor Data Mover:
//     2D D#: tile_dim0=1, tile_dim1=1519, stride0=8 elems, data_size=4B
//   S(16x32) = Q(16x32) Kt            -> 2x wmma
//   online softmax (16-lane shfl_xor butterflies per the C-matrix layout)
//   O(16x32) += P(16x32) V            -> 2x wmma (P transposed via LDS)
// rel-bias computed analytically: idx = base(i) - (tj*49+vj) + 759
// ---------------------------------------------------------------------------
__global__ __launch_bounds__(32)
void attn_kernel(const _Float16* __restrict__ qh,
                 const _Float16* __restrict__ kh,
                 const _Float16* __restrict__ vT,
                 const float* __restrict__ pos_tab,
                 _Float16* __restrict__ ao) {
  __shared__ __align__(16) float    lds_pos[1536];
  __shared__ __align__(16) _Float16 ldsP[16 * 40];

  const int lane = threadIdx.x;
  const int ln16 = lane & 15;
  const int hi   = lane >> 4;
  const int m0   = blockIdx.x * 16;   // row tile (25 tiles)
  const int h    = blockIdx.y;
  const int b    = blockIdx.z;

  // ---- TDM: DMA this head's bias column into LDS (single-wave block =>
  //      exactly one TENSOR_LOAD_TO_LDS issued; tracked by TENSORcnt) ----
  {
    const unsigned long long gaddr =
        (unsigned long long)(const void*)(pos_tab + h);  // tile start (bytes)
    u32x4 g0;
    g0[0] = 1u;                                   // count=1, user descriptor
    g0[1] = lds_byte_addr(lds_pos);               // LDS dest byte address
    g0[2] = (unsigned)(gaddr & 0xffffffffu);      // global_addr[31:0]
    g0[3] = (unsigned)((gaddr >> 32) & 0x01ffffffu) | (2u << 30);  // + type=2
    i32x8 g1;
    g1[0] = (2 << 16);          // workgroup_mask=0, data_size=2 (4 bytes)
    g1[1] = (8 << 16);          // tensor_dim0 = 8 (row of pos_tab), low bits
    g1[2] = (NPOS << 16);       // tensor_dim1 = 1519 (low16) | dim0 hi16 = 0
    g1[3] = (1 << 16);          // tile_dim0 = 1 | tensor_dim1 hi16 = 0
    g1[4] = NPOS;               // tile_dim2 = 0 | tile_dim1 = 1519
    g1[5] = 8;                  // tensor_dim0_stride = 8 elements (32 B)
    g1[6] = 0;                  // stride0 hi | stride1 lo
    g1[7] = 0;                  // stride1 hi
    i32x4 g2 = {};              // 2D tensor: groups 2/3 zero
    i32x4 g3 = {};
    i32x8 g4 = {};              // extra group (clang-23 6-arg form), unused
    __builtin_amdgcn_tensor_load_to_lds(g0, g1, g2, g3, g4, 0);
  }

  // Q fragment straight from global in contiguous k order (lanes 16-31 hold K=16..31)
  const size_t qbase = ((size_t)(b * HEADS + h) * NTOK + (m0 + ln16)) * HD;
  const int kbA = hi ? 16 : 0;
  const v16h qfrag = concat8(*(const v8h*)(qh + qbase + kbA),
                             *(const v8h*)(qh + qbase + kbA + 8));

  int basecode[8];
#pragma unroll
  for (int r = 0; r < 8; ++r) {
    const int i = m0 + r + hi * 8;                 // row this (lane,reg) owns
    basecode[r] = (i / 25) * 49 + (i % 25) + 15 * 49 + 24;
  }

  v8f accO0 = {}, accO1 = {};
  float mrow[8], lrow[8];
#pragma unroll
  for (int r = 0; r < 8; ++r) { mrow[r] = -3.0e38f; lrow[r] = 0.f; }

  const size_t khead = (size_t)(b * HEADS + h) * NTOK * HD;
  const size_t vhead = (size_t)(b * HEADS + h) * HD * NPAD;
  const v8f zero8 = {};

  __builtin_amdgcn_s_wait_tensorcnt(0);   // pos column resident in LDS
  __syncthreads();

  for (int j0 = 0; j0 < NTOK; j0 += 32) {
    float s[2][8];
#pragma unroll
    for (int jj = 0; jj < 2; ++jj) {
      const int col = j0 + jj * 16 + ln16;
      const bool valid = col < NTOK;
      v16h kfrag = {};
      if (valid) {
        const _Float16* kp = kh + khead + (size_t)col * HD + (hi ? 16 : 0);
        kfrag = concat8(*(const v8h*)kp, *(const v8h*)(kp + 8));
      }
      const v8f sv = wmma_f16(qfrag, kfrag, zero8);
      const int jcode = (col / 25) * 49 + (col % 25);
#pragma unroll
      for (int r = 0; r < 8; ++r)
        s[jj][r] = valid ? sv[r] + lds_pos[basecode[r] - jcode] : -1.0e30f;
    }

    // online softmax per row (row lives in one 16-lane half of the wave)
#pragma unroll
    for (int r = 0; r < 8; ++r) {
      float cmax = fmaxf(s[0][r], s[1][r]);
#pragma unroll
      for (int msk = 1; msk < 16; msk <<= 1)
        cmax = fmaxf(cmax, __shfl_xor(cmax, msk, 32));
      const float mnew  = fmaxf(mrow[r], cmax);
      const float alpha = __expf(mrow[r] - mnew);
      const float p0 = __expf(s[0][r] - mnew);
      const float p1 = __expf(s[1][r] - mnew);
      float csum = p0 + p1;
#pragma unroll
      for (int msk = 1; msk < 16; msk <<= 1)
        csum += __shfl_xor(csum, msk, 32);
      lrow[r] = lrow[r] * alpha + csum;
      mrow[r] = mnew;
      accO0[r] *= alpha;
      accO1[r] *= alpha;
      const int M = r + hi * 8;                     // C layout -> P tile row
      ldsP[M * 40 + ln16]      = (_Float16)p0;
      ldsP[M * 40 + 16 + ln16] = (_Float16)p1;
    }
    __syncthreads();

    // P as A-fragment, V^T rows as B-fragments (contiguous in kv)
    const v16h pfrag = concat8(*(const v8h*)&ldsP[ln16 * 40 + kbA],
                               *(const v8h*)&ldsP[ln16 * 40 + kbA + 8]);
    const _Float16* vb0 = vT + vhead + (size_t)ln16 * NPAD + j0 + (hi ? 16 : 0);
    const _Float16* vb1 = vb0 + (size_t)16 * NPAD;
    const v16h vf0 = concat8(*(const v8h*)vb0, *(const v8h*)(vb0 + 8));
    const v16h vf1 = concat8(*(const v8h*)vb1, *(const v8h*)(vb1 + 8));
    accO0 = wmma_f16(pfrag, vf0, accO0);
    accO1 = wmma_f16(pfrag, vf1, accO1);
    __syncthreads();
  }

  // normalize and store [B,N,H*32] as f16 for the projection GEMM
#pragma unroll
  for (int r = 0; r < 8; ++r) {
    const float inv = 1.0f / lrow[r];
    const int i = m0 + r + hi * 8;
    const size_t obase = ((size_t)b * NTOK + i) * DIM + h * HD;
    ao[obase + ln16]      = (_Float16)(accO0[r] * inv);
    ao[obase + 16 + ln16] = (_Float16)(accO1[r] * inv);
  }
}

// ---------------------------------------------------------------------------
// Kernel 4: out = ao @ proj_w + proj_b   (51200 x 256 x 256, f16 A, f32 out)
// ---------------------------------------------------------------------------
__global__ __launch_bounds__(256)
void proj_gemm_kernel(const _Float16* __restrict__ ao,
                      const float* __restrict__ w,
                      const float* __restrict__ bias,
                      float* __restrict__ out) {
  __shared__ __align__(16) _Float16 ldsA[128 * 40];
  __shared__ __align__(16) _Float16 ldsB[64 * 40];

  const int t    = threadIdx.x;
  const int lane = t & 31;
  const int wave = t >> 5;
  const int ln16 = lane & 15;
  const int hi   = lane >> 4;
  const int m0   = blockIdx.x * 128;
  const int n0   = blockIdx.y * 64;

  v8f acc[4] = {};

  for (int kk = 0; kk < DIM; kk += 32) {
#pragma unroll
    for (int u = 0; u < 2; ++u) {
      int c   = u * 256 + t;        // 512 v8h chunks
      int row = c >> 2;
      int c8  = (c & 3) * 8;
      *(v8h*)&ldsA[row * 40 + c8] =
          *(const v8h*)(ao + (size_t)(m0 + row) * DIM + kk + c8);
    }
    {
      int col = t & 63;
      int kg  = t >> 6;
      v8h pk;
#pragma unroll
      for (int k8 = 0; k8 < 8; ++k8)
        pk[k8] = (_Float16)w[(size_t)(kk + kg * 8 + k8) * DIM + n0 + col];
      *(v8h*)&ldsB[col * 40 + kg * 8] = pk;
    }
    __syncthreads();

    const int arow = wave * 16 + ln16;
    const int kbA  = hi ? 16 : 0;
    v16h afrag = concat8(*(const v8h*)&ldsA[arow * 40 + kbA],
                         *(const v8h*)&ldsA[arow * 40 + kbA + 8]);
    const int kbB = hi ? 16 : 0;
#pragma unroll
    for (int j = 0; j < 4; ++j) {
      const int col = j * 16 + ln16;
      v16h bfrag = concat8(*(const v8h*)&ldsB[col * 40 + kbB],
                           *(const v8h*)&ldsB[col * 40 + kbB + 8]);
      acc[j] = wmma_f16(afrag, bfrag, acc[j]);
    }
    __syncthreads();
  }

#pragma unroll
  for (int j = 0; j < 4; ++j) {
    const int col = n0 + j * 16 + ln16;
    const float bv = bias[col];
#pragma unroll
    for (int r = 0; r < 8; ++r) {
      const int g = m0 + wave * 16 + r + hi * 8;
      out[(size_t)g * DIM + col] = acc[j][r] + bv;
    }
  }
}

// ---------------------------------------------------------------------------
// Host launcher
// ---------------------------------------------------------------------------
extern "C" void kernel_launch(void* const* d_in, const int* in_sizes, int n_in,
                              void* d_out, int out_size, void* d_ws,
                              size_t ws_size, hipStream_t stream) {
  const float* x      = (const float*)d_in[0];
  const float* qkv_w  = (const float*)d_in[1];
  const float* qkv_b  = (const float*)d_in[2];
  const float* proj_w = (const float*)d_in[3];
  const float* proj_b = (const float*)d_in[4];
  const float* pp_w   = (const float*)d_in[5];
  const float* pp_b   = (const float*)d_in[6];
  const float* ln1g   = (const float*)d_in[7];
  const float* ln1b   = (const float*)d_in[8];
  const float* p1w    = (const float*)d_in[9];
  const float* p1b    = (const float*)d_in[10];
  const float* ln2g   = (const float*)d_in[11];
  const float* ln2b   = (const float*)d_in[12];
  const float* p2w    = (const float*)d_in[13];
  const float* p2b    = (const float*)d_in[14];
  const float* ln3g   = (const float*)d_in[15];
  const float* ln3b   = (const float*)d_in[16];
  const float* p3w    = (const float*)d_in[17];
  const float* p3b    = (const float*)d_in[18];

  unsigned char* ws = (unsigned char*)d_ws;
  const size_t QKV_BYTES = (size_t)BATCH * HEADS * NTOK * HD * 2;   // 26,214,400
  const size_t VT_BYTES  = (size_t)BATCH * HEADS * HD * NPAD * 2;   // 27,262,976
  const size_t AO_BYTES  = (size_t)BATCH * NTOK * DIM * 2;          // 26,214,400
  _Float16* qh      = (_Float16*)(ws);
  _Float16* kh      = (_Float16*)(ws + QKV_BYTES);
  _Float16* vT      = (_Float16*)(ws + 2 * QKV_BYTES);
  _Float16* ao      = (_Float16*)(ws + 2 * QKV_BYTES + VT_BYTES);
  float*    pos_tab = (float*)   (ws + 2 * QKV_BYTES + VT_BYTES + AO_BYTES);

  qkv_gemm_kernel<<<dim3(400, 12), 256, 0, stream>>>(x, qkv_w, qkv_b, qh, kh, vT);
  posmlp_kernel<<<dim3((NPOS + 255) / 256), 256, 0, stream>>>(
      pp_w, pp_b, ln1g, ln1b, p1w, p1b, ln2g, ln2b, p2w, p2b, ln3g, ln3b, p3w,
      p3b, pos_tab);
  attn_kernel<<<dim3(25, HEADS, BATCH), 32, 0, stream>>>(qh, kh, vT, pos_tab, ao);
  proj_gemm_kernel<<<dim3(400, 4), 256, 0, stream>>>(ao, proj_w, proj_b,
                                                     (float*)d_out);
}
